// Attention_9749575762765
// MI455X (gfx1250) — matrix operand order repaired and checked by reading, not compiled, not run on hardware
//
#include <hip/hip_runtime.h>

typedef __attribute__((ext_vector_type(16))) __bf16          v16bf;
typedef __attribute__((ext_vector_type(8)))  float           v8f;
typedef __attribute__((ext_vector_type(8)))  unsigned short  us8;
typedef __attribute__((ext_vector_type(16))) unsigned short  us16;

#define M_TOT 16384   // B*S
#define N_TOT 2048    // DIM
#define K_TOT 2048    // DIM
#define S_LEN 4096
#define HD    128

#define BM 128
#define BN 128
#define BK 64
#define LDA (BK + 8)  // 72 halves = 144B row stride -> conflict-free b128 frag loads

__device__ __forceinline__ unsigned short f2bf(float f) {
  unsigned int u = __float_as_uint(f);
  u += 0x7FFFu + ((u >> 16) & 1u);   // round-to-nearest-even
  return (unsigned short)(u >> 16);
}

// ---------------- Kernel 1: X f32 -> bf16 ----------------
__global__ void cvt_x(const float* __restrict__ X, unsigned short* __restrict__ Xb) {
  int i = blockIdx.x * blockDim.x + threadIdx.x;   // one us8 (8 elems) per thread
  const float4* p = (const float4*)X + (size_t)i * 2;
  float4 a = p[0], b = p[1];
  us8 o;
  o[0] = f2bf(a.x); o[1] = f2bf(a.y); o[2] = f2bf(a.z); o[3] = f2bf(a.w);
  o[4] = f2bf(b.x); o[5] = f2bf(b.y); o[6] = f2bf(b.z); o[7] = f2bf(b.w);
  *((us8*)Xb + i) = o;
}

// ------------- Kernel 2: W f32 -> bf16 W^T (N-major) -------------
__global__ void cvt_wt(const float* __restrict__ W0, const float* __restrict__ W1,
                       const float* __restrict__ W2, unsigned short* __restrict__ WtAll) {
  const float* W = (blockIdx.z == 0) ? W0 : ((blockIdx.z == 1) ? W1 : W2);
  unsigned short* O = WtAll + (size_t)blockIdx.z * K_TOT * N_TOT;
  __shared__ float t[32][33];
  int n0 = blockIdx.x * 32, k0 = blockIdx.y * 32;
  int tx = threadIdx.x, ty = threadIdx.y;
#pragma unroll
  for (int i = 0; i < 32; i += 8)
    t[ty + i][tx] = W[(size_t)(k0 + ty + i) * N_TOT + n0 + tx];
  __syncthreads();
#pragma unroll
  for (int i = 0; i < 32; i += 8)
    O[(size_t)(n0 + ty + i) * K_TOT + k0 + tx] = f2bf(t[tx][ty + i]);
}

// issue 4 async b128 global->LDS copies covering a 128x64 bf16 tile slice
__device__ __forceinline__ void async_tile_load(unsigned short* lds,
                                                const unsigned short* g,
                                                int lr, int lc) {
#pragma unroll
  for (int rr = 0; rr < BM; rr += 32) {
    int r = lr + rr;
    unsigned ldsoff = (unsigned)(uintptr_t)(&lds[r * LDA + lc]);
    const void* gp = (const void*)(g + (size_t)r * K_TOT + lc);
    asm volatile("global_load_async_to_lds_b128 %0, %1, off"
                 :: "v"(ldsoff), "v"(gp)
                 : "memory");
  }
}

// ------------- Kernel 3: tiled WMMA GEMM (async double-buffer) + RoPE -------------
__global__ __launch_bounds__(256) void gemm_rope(
    const unsigned short* __restrict__ Xb,
    const unsigned short* __restrict__ WtAll,
    const float* __restrict__ fc, const float* __restrict__ fs,
    float* __restrict__ outAll)
{
  const int mat = blockIdx.z;                       // 0=Q, 1=K, 2=V
  const unsigned short* Wt = WtAll + (size_t)mat * K_TOT * N_TOT;
  float* Out = outAll + (size_t)mat * M_TOT * N_TOT;

  const int mBase = blockIdx.y * BM;
  const int nBase = blockIdx.x * BN;

  __shared__ unsigned short As[2][BM * LDA];
  __shared__ unsigned short Bs[2][BN * LDA];

  const int tid  = threadIdx.x;
  const int lane = tid & 31;
  const int wid  = tid >> 5;
  const int wm   = wid & 3;     // wave M offset = wm*32
  const int wn   = wid >> 2;    // wave N offset = wn*64
  const int l16  = lane & 15;
  const int half = lane >> 4;

  v8f acc[2][4];
  {
    v8f z = {};
#pragma unroll
    for (int i = 0; i < 2; i++)
#pragma unroll
      for (int j = 0; j < 4; j++) acc[i][j] = z;
  }

  const int lr = tid >> 3;         // 0..31 : staging row
  const int lc = (tid & 7) * 8;    // 0..56 : staging col (8 halves = b128)

  const unsigned short* Ag = Xb + (size_t)mBase * K_TOT;
  const unsigned short* Bg = Wt + (size_t)nBase * K_TOT;

  // prologue: async-stage tile 0 into buffer 0 (8 ops/thread)
  async_tile_load(As[0], Ag, lr, lc);
  async_tile_load(Bs[0], Bg, lr, lc);

  const int NK = K_TOT / BK;       // 32
  for (int it = 0; it < NK; ++it) {
    const int b = it & 1;
    if (it + 1 < NK) {
      // overlap: stage next tile while we compute on current
      async_tile_load(As[b ^ 1], Ag + (it + 1) * BK, lr, lc);
      async_tile_load(Bs[b ^ 1], Bg + (it + 1) * BK, lr, lc);
      // async ops complete in order: <=8 outstanding means current tile landed
      asm volatile("s_wait_asynccnt 0x8" ::: "memory");
    } else {
      asm volatile("s_wait_asynccnt 0x0" ::: "memory");
    }
    __syncthreads();

#pragma unroll
    for (int kk = 0; kk < BK; kk += 32) {
      v16bf a[2], bf[4];
      // A fragments: ISA 16-bit A layout -> two b128 chunks per lane
#pragma unroll
      for (int i = 0; i < 2; i++) {
        int row = wm * 32 + i * 16 + l16;
        const unsigned short* p = &As[b][row * LDA + kk + half * 8];
        us8 lo = *(const us8*)p;
        us8 hi = *(const us8*)(p + 16);
        us16 c = __builtin_shufflevector(lo, hi, 0,1,2,3,4,5,6,7,8,9,10,11,12,13,14,15);
        a[i] = __builtin_bit_cast(v16bf, c);
      }
      // B fragments: lane holds column N=l16, K = half*8 + {0..7, 16..23}
#pragma unroll
      for (int j = 0; j < 4; j++) {
        int row = wn * 64 + j * 16 + l16;
        const unsigned short* p = &Bs[b][row * LDA + kk + half * 8];
        us8 lo = *(const us8*)p;
        us8 hi = *(const us8*)(p + 16);
        us16 c = __builtin_shufflevector(lo, hi, 0,1,2,3,4,5,6,7,8,9,10,11,12,13,14,15);
        bf[j] = __builtin_bit_cast(v16bf, c);
      }
#pragma unroll
      for (int i = 0; i < 2; i++)
#pragma unroll
        for (int j = 0; j < 4; j++)
          acc[i][j] = __builtin_amdgcn_wmma_f32_16x16x32_bf16(
              false, a[i], false, bf[j], (short)0, acc[i][j], false, false);
    }
    __syncthreads();   // all waves done reading buf b before it is overwritten
  }

  // ---- epilogue: interleaved RoPE for Q,K; plain store for V ----
  const bool doRope = (mat < 2);
#pragma unroll
  for (int i = 0; i < 2; i++) {
    int mSub = mBase + wm * 32 + i * 16 + half * 8;   // C frag: lanes>=16 -> M+8
#pragma unroll
    for (int j = 0; j < 4; j++) {
      int gn = nBase + wn * 64 + j * 16 + l16;        // C frag: N = lane%16
      int d  = gn & (HD - 1);
#pragma unroll
      for (int r = 0; r < 8; r++) {
        int   gm = mSub + r;
        float v  = acc[i][j][r];
        float res;
        if (doRope) {
          int   s    = gm & (S_LEN - 1);
          float part = __shfl_xor(v, 1, 32);          // interleaved pair partner
          float cs   = fc[s * HD + d];
          float sn   = fs[s * HD + d];
          // even d: v*cos - pair*sin ; odd d: v*cos + pair*sin
          res = v * cs + ((l16 & 1) ? part : -part) * sn;
        } else {
          res = v;
        }
        Out[(size_t)gm * N_TOT + gn] = res;
      }
    }
  }
}

extern "C" void kernel_launch(void* const* d_in, const int* in_sizes, int n_in,
                              void* d_out, int out_size, void* d_ws, size_t ws_size,
                              hipStream_t stream) {
  const float* X  = (const float*)d_in[0];
  const float* fc = (const float*)d_in[1];
  const float* fs = (const float*)d_in[2];
  // d_in[3] attention_mask: unused by the reference math
  const float* Wq = (const float*)d_in[4];
  const float* Wk = (const float*)d_in[5];
  const float* Wv = (const float*)d_in[6];

  unsigned short* Xb = (unsigned short*)d_ws;                       // 16384*2048 bf16
  unsigned short* Wt = Xb + (size_t)M_TOT * K_TOT;                  // 3 * 2048*2048 bf16

  // X -> bf16 : 33,554,432 elems / 8 per thread = 4,194,304 threads
  cvt_x<<<16384, 256, 0, stream>>>(X, Xb);

  // W -> bf16 W^T : 32x32 tiles
  cvt_wt<<<dim3(N_TOT / 32, K_TOT / 32, 3), dim3(32, 8, 1), 0, stream>>>(Wq, Wk, Wv, Wt);

  // fused GEMM + RoPE (async double-buffered)
  gemm_rope<<<dim3(N_TOT / BN, M_TOT / BM, 3), 256, 0, stream>>>(
      Xb, Wt, fc, fs, (float*)d_out);
}
